// ModuleCorrelation_283467841943
// MI455X (gfx1250) — compile-verified, matrix operand order checked by reading, not run on hardware
//
#include <hip/hip_runtime.h>

typedef __attribute__((ext_vector_type(16))) _Float16 v16h;
typedef __attribute__((ext_vector_type(8)))  float    v8f;

#define BB   4
#define CC   128
#define HHH  128
#define WWW  224
#define PADD 4
#define HT   4      // h-rows per block == waves per block
#define ND   81

// K index inside a 16x32 f16 A/B fragment: element i of the v16h, per the
// CDNA5 ISA layout (VGPR0..3 hold K=0..7 / K=8..15 by lane half, VGPR4..7 add +16)
__device__ __forceinline__ int kmap(int i) { return i + ((i >= 8) ? 8 : 0); }

// Compute one vertical shift's two N-tiles (8 WMMAs). EDGE=true adds the
// zero-masking (cndmask) path for tiles touching the padded border.
template <bool EDGE>
__device__ __forceinline__ void corr_tiles(const float* __restrict__ sbase,
                                           const v16h (&afrag)[4],
                                           int m, int kbase, int w0, bool rowok,
                                           size_t plane, v8f (&acc)[2])
{
    #pragma unroll
    for (int t = 0; t < 2; ++t) {
        const int n    = t * 16 + m;              // column of S tile
        // Columns n>23 never reach a useful band (o = N-M would exceed 8):
        // clamp them onto column 23 so their loads alias an already-needed line.
        const int neff = (n > 23) ? 23 : n;
        const int wcol = w0 - PADD + neff;        // absolute w in second

        bool ok;
        int  wc;
        if (EDGE) {
            ok = rowok && (wcol >= 0) && (wcol < WWW);
            wc = ok ? wcol : 0;                   // clamped -> always valid
        } else {
            ok = true;
            wc = wcol;                            // interior: always valid
        }
        const float* sb = sbase + (size_t)kbase * plane + wc;

        // Unconditional loads: all 64 in flight behind staggered waits.
        float tmp[64];
        #pragma unroll
        for (int chunk = 0; chunk < 4; ++chunk)
            #pragma unroll
            for (int i = 0; i < 16; ++i)
                tmp[chunk * 16 + i] =
                    sb[(size_t)(chunk * 32 + kmap(i)) * plane];

        #pragma unroll
        for (int chunk = 0; chunk < 4; ++chunk) {
            v16h bfrag;
            #pragma unroll
            for (int i = 0; i < 16; ++i) {
                float v = tmp[chunk * 16 + i];
                if (EDGE) v = ok ? v : 0.0f;      // v_cndmask, no branch
                bfrag[i] = (_Float16)v;
            }
            acc[t] = __builtin_amdgcn_wmma_f32_16x16x32_f16(
                false, afrag[chunk], false, bfrag, (short)0, acc[t],
                false, false);
        }
    }
}

__global__ __launch_bounds__(32 * HT)
void corr81_wmma_f16(const float* __restrict__ first,
                     const float* __restrict__ second,
                     float* __restrict__ out)
{
    const int lane = threadIdx.x & 31;
    const int wave = threadIdx.x >> 5;

    const int wt = blockIdx.x;              // 0..13  (W / 16)
    const int h  = blockIdx.y * HT + wave;  // 0..127
    const int b  = blockIdx.z;              // 0..3
    const int w0 = wt * 16;

    const int m     = lane & 15;            // row (A) / col-within-tile (B)
    const int halfw = lane >> 4;            // lane half selects Kbase
    const int kbase = halfw * 8;

    __shared__ float outstage[HT][9 * 16];  // [o][M] per wave

    const size_t plane = (size_t)HHH * WWW; // stride between channels

    // Tile is w-interior iff the 24-column window w0-4 .. w0+19 is in range.
    const bool winterior = (w0 >= PADD) && (w0 + 19 < WWW);

    // ---------- A fragments: first[b, :, h, w0 + m] (built once) ----------
    const float* fbase = first + ((size_t)(b * CC) * HHH + h) * WWW + (w0 + m);
    const float* fb2   = fbase + (size_t)kbase * plane;
    v16h afrag[4];
    {
        float tmpa[64];
        #pragma unroll
        for (int chunk = 0; chunk < 4; ++chunk)
            #pragma unroll
            for (int i = 0; i < 16; ++i)
                tmpa[chunk * 16 + i] = fb2[(size_t)(chunk * 32 + kmap(i)) * plane];
        #pragma unroll
        for (int chunk = 0; chunk < 4; ++chunk)
            #pragma unroll
            for (int i = 0; i < 16; ++i)
                afrag[chunk][i] = (_Float16)tmpa[chunk * 16 + i];
    }

    const float scale = 1.0f / (float)CC;

    for (int p = 0; p < 9; ++p) {
        const int  h2    = h + p - PADD;
        const bool rowok = (h2 >= 0) && (h2 < HHH);
        const float* sbase =
            second + ((size_t)(b * CC) * HHH + (rowok ? h2 : 0)) * WWW;

        // Prefetch hint for next vertical shift's row (L2-resident anyway).
        if (p < 8) {
            const int h2n = h + p + 1 - PADD;
            if (h2n >= 0 && h2n < HHH) {
                const float* pf =
                    second + ((size_t)(b * CC + lane * 4) * HHH + h2n) * WWW + w0;
                __builtin_prefetch(pf, 0, 1);
            }
        }

        v8f acc[2];
        acc[0] = (v8f){0.f, 0.f, 0.f, 0.f, 0.f, 0.f, 0.f, 0.f};
        acc[1] = (v8f){0.f, 0.f, 0.f, 0.f, 0.f, 0.f, 0.f, 0.f};

        // Wave-uniform branch: mask-free fast path for interior tiles.
        if (rowok && winterior)
            corr_tiles<false>(sbase, afrag, m, kbase, w0, rowok, plane, acc);
        else
            corr_tiles<true >(sbase, afrag, m, kbase, w0, rowok, plane, acc);

        // ---- extract diagonal bands: out(w0+M, o) lives at D[M, N=M+o] ----
        #pragma unroll
        for (int t = 0; t < 2; ++t) {
            #pragma unroll
            for (int r = 0; r < 8; ++r) {
                const int M = r + 8 * halfw;      // D-row  (w offset in tile)
                const int N = m + 16 * t;         // D-col
                const int o = N - M;              // horizontal displacement
                if (o >= 0 && o <= 8)
                    outstage[wave][o * 16 + M] = acc[t][r] * scale;
            }
        }

        // ---- coalesced writeback: 9 rows of 16 consecutive floats ----
        float* obase = out + (((size_t)(b * ND) + p * 9) * HHH + h) * WWW + w0;
        for (int t2 = lane; t2 < 144; t2 += 32) {
            const int o  = t2 >> 4;
            const int mm = t2 & 15;
            obase[(size_t)o * plane + mm] = outstage[wave][t2];
        }
    }
}

extern "C" void kernel_launch(void* const* d_in, const int* in_sizes, int n_in,
                              void* d_out, int out_size, void* d_ws, size_t ws_size,
                              hipStream_t stream)
{
    (void)in_sizes; (void)n_in; (void)d_ws; (void)ws_size; (void)out_size;
    const float* first  = (const float*)d_in[0];
    const float* second = (const float*)d_in[1];
    float*       out    = (float*)d_out;

    dim3 grid(WWW / 16, HHH / HT, BB);   // (14, 32, 4)
    dim3 block(32 * HT);                 // 4 wave32 waves
    hipLaunchKernelGGL(corr81_wmma_f16, grid, block, 0, stream,
                       first, second, out);
}
